// SimplifiedSelfAttention_43069932044296
// MI455X (gfx1250) — matrix operand order repaired and checked by reading, not compiled, not run on hardware
//
#include <hip/hip_runtime.h>
#include <hip/hip_bf16.h>
#include <stdint.h>

typedef __attribute__((ext_vector_type(16))) __bf16 v16bf;
typedef __attribute__((ext_vector_type(8)))  __bf16 v8bf;
typedef __attribute__((ext_vector_type(8)))  float  v8f;

#define NB   4
#define LQ   2048
#define NH   8
#define HD   64
#define FEA  512   // NH*HD
#define KT   32    // keys per streamed tile
#define NT   (LQ / KT)

union V16U { v16bf v; v8bf h[2]; };

static __device__ inline v8f wmma_bf16(v16bf a, v16bf b, v8f c) {
    // D = A(16x32 bf16) * B(32x16 bf16) + C(16x16 f32)
    return __builtin_amdgcn_wmma_f32_16x16x32_bf16(false, a, false, b, (short)0, c, false, false);
}

// ---- CDNA5 async global->LDS path (guarded; falls back to sync staging) ----
#if __has_builtin(__builtin_amdgcn_global_load_async_to_lds_b128)
#define SA_HAVE_ASYNC_LDS 1
#else
#define SA_HAVE_ASYNC_LDS 0
#endif

#if __has_builtin(__builtin_amdgcn_s_wait_asynccnt)
#define SA_WAIT_ASYNC(n) __builtin_amdgcn_s_wait_asynccnt(n)
#else
#define SA_WAIT_ASYNC(n) asm volatile("s_wait_asynccnt %0" :: "n"(n) : "memory")
#endif

// Builtin takes pointers to 16-byte int vectors in AS1 (global) / AS3 (LDS).
typedef int v4i __attribute__((vector_size(16)));
typedef __attribute__((address_space(1))) v4i* as1_v4i_p;
typedef __attribute__((address_space(3))) v4i* as3_v4i_p;

static __device__ __forceinline__ void async_cp16(const __bf16* g, __bf16* s) {
#if SA_HAVE_ASYNC_LDS
    __builtin_amdgcn_global_load_async_to_lds_b128(
        (as1_v4i_p)(void*)g, (as3_v4i_p)(void*)s, 0, 0);
#else
    *(v8bf*)s = *(const v8bf*)g;
#endif
}

// ---------------------------------------------------------------------------
// Kernel 1: per-head projections. One block (64 threads) per (n,l,h) row.
// Emits Qp,Kp as [n][h][l][64] bf16 and Vt as [n][h][64][l] bf16 (transposed
// so the P*V B-operand is a contiguous 16-half run per lane).
// ---------------------------------------------------------------------------
__global__ __launch_bounds__(64)
void sa_proj_kernel(const float* __restrict__ q,
                    const float* __restrict__ Wv,
                    const float* __restrict__ Wk,
                    const float* __restrict__ Wq,
                    __bf16* __restrict__ Qp, __bf16* __restrict__ Kp,
                    __bf16* __restrict__ Vt) {
    __shared__ float xs[HD];
    const int row = blockIdx.x;                 // (n*L + l)*H + h
    const int h   = row % NH;
    const int l   = (row / NH) % LQ;
    const int n   = row / (NH * LQ);
    const int e   = threadIdx.x;                // output column 0..63

    const float* x = q + ((size_t)(n * LQ + l)) * FEA + h * HD;
    xs[e] = x[e];
    __syncthreads();

    float av = 0.f, ak = 0.f, aq = 0.f;
    #pragma unroll 8
    for (int d = 0; d < HD; ++d) {
        const float xv = xs[d];
        av += xv * Wv[d * HD + e];
        ak += xv * Wk[d * HD + e];
        aq += xv * Wq[d * HD + e];
    }
    const size_t nh = (size_t)(n * NH + h);
    const size_t ro = (nh * LQ + l) * HD + e;
    Qp[ro] = (__bf16)aq;
    Kp[ro] = (__bf16)ak;
    Vt[(nh * HD + e) * LQ + l] = (__bf16)av;
}

// ---------------------------------------------------------------------------
// Kernel 2: Wo [512][64] f32 -> Wot [64][512] bf16 (B-operand friendly).
// ---------------------------------------------------------------------------
__global__ __launch_bounds__(256)
void sa_wot_kernel(const float* __restrict__ Wo, __bf16* __restrict__ Wot) {
    const int idx = blockIdx.x * 256 + threadIdx.x;  // 0..32767 = e*512 + d
    const int e = idx >> 9;
    const int d = idx & 511;
    Wot[idx] = (__bf16)Wo[(size_t)d * HD + e];
}

// ---------------------------------------------------------------------------
// Kernel 3: fused attention core. 256 threads = 8 waves; each wave owns a
// 16-query x 64-d output tile of one (n,h). K/V tiles are staged into LDS
// ONCE per block (async, double-buffered) instead of 8x per-wave from L2:
//   S(16x32) via 4 wmma, P = relu(S)^2, LDS round-trip to A-layout,
//   out += P*V via 4 wmma. Scores never touch HBM.
// ---------------------------------------------------------------------------
static __device__ __forceinline__ void sa_stage(const __bf16* __restrict__ Kbase,
                                                const __bf16* __restrict__ Vbase,
                                                __bf16* Ks, __bf16* Vs,
                                                int tid, int kbase) {
    // K tile: 32 keys x 64 d, contiguous 4 KB; each thread moves one 16B chunk
    async_cp16(Kbase + (size_t)kbase * HD + tid * 8, Ks + tid * 8);
    // V tile: 64 d-rows x 32 keys (row stride LQ in global, 32 in LDS)
    const int d = tid >> 2, c = tid & 3;
    async_cp16(Vbase + (size_t)d * LQ + kbase + c * 8, Vs + d * KT + c * 8);
}

__global__ __launch_bounds__(256)
void sa_attn_kernel(const __bf16* __restrict__ Qp,
                    const __bf16* __restrict__ Kp,
                    const __bf16* __restrict__ Vt,
                    __bf16* __restrict__ Ab) {
    __shared__ __align__(16) __bf16 Kt[2][KT * HD];   // 2 x 4KB
    __shared__ __align__(16) __bf16 Vl[2][HD * KT];   // 2 x 4KB, [d][key]
    __shared__ __align__(16) __bf16 Pt[8][16 * 32];   // per-wave 1KB P scratch

    const int tid  = threadIdx.x;
    const int wave = tid >> 5;
    const int lane = tid & 31;
    const int nn   = lane & 15;                 // column / row-within-tile id
    const int h2   = lane >> 4;                 // half-wave select

    const int nh   = blockIdx.x >> 4;           // 0..31  (n*8 + h)
    const int qblk = blockIdx.x & 15;
    const int q0   = qblk * 128 + wave * 16;    // first query row of this wave

    const __bf16* Qbase = Qp + (size_t)nh * LQ * HD;
    const __bf16* Kbase = Kp + (size_t)nh * LQ * HD;
    const __bf16* Vbase = Vt + (size_t)nh * HD * LQ;

    // Q A-fragments (held for the whole key loop): d 0..31 and 32..63
    const __bf16* qrow = Qbase + (size_t)(q0 + nn) * HD;
    V16U qa0, qa1;
    qa0.h[0] = *(const v8bf*)(qrow +      h2 * 16);
    qa0.h[1] = *(const v8bf*)(qrow +  8 + h2 * 16);
    qa1.h[0] = *(const v8bf*)(qrow + 32 + h2 * 16);
    qa1.h[1] = *(const v8bf*)(qrow + 40 + h2 * 16);

    v8f z = {};
    v8f acc[4] = {z, z, z, z};

    // prime the pipeline with tile 0
    sa_stage(Kbase, Vbase, Kt[0], Vl[0], tid, 0);

    for (int kt = 0; kt < NT; ++kt) {
        const int kbase = kt * KT;
        if (kt + 1 < NT)
            sa_stage(Kbase, Vbase, Kt[(kt + 1) & 1], Vl[(kt + 1) & 1],
                     tid, kbase + KT);
#if SA_HAVE_ASYNC_LDS
        if (kt + 1 < NT) { SA_WAIT_ASYNC(2); } else { SA_WAIT_ASYNC(0); }
#endif
        __syncthreads();                        // tile kt resident for all waves

        const __bf16* Ks = Kt[kt & 1];
        const __bf16* Vs = Vl[kt & 1];

        // --- scores: two 16x16 tiles (keys +0..15, +16..31), K-dim 64 in 2 steps
        #pragma unroll
        for (int s = 0; s < 2; ++s) {
            const __bf16* krow = Ks + (s * 16 + nn) * HD;
            const v16bf b0 = *(const v16bf*)(krow +      h2 * 16);   // d 0..31
            const v16bf b1 = *(const v16bf*)(krow + 32 + h2 * 16);   // d 32..63
            v8f sc = {};
            sc = wmma_bf16(qa0.v, b0, sc);
            sc = wmma_bf16(qa1.v, b1, sc);
            // relu^2 -> bf16, scatter D-layout (lane=col nn, rows r+8*h2)
            #pragma unroll
            for (int r = 0; r < 8; ++r) {
                float t = sc[r] > 0.f ? sc[r] : 0.f;
                Pt[wave][(r + 8 * h2) * 32 + s * 16 + nn] = (__bf16)(t * t);
            }
        }

        // --- gather P back as a 16x32 bf16 A-fragment
        V16U pa;
        pa.h[0] = *(const v8bf*)&Pt[wave][nn * 32 +      h2 * 16];
        pa.h[1] = *(const v8bf*)&Pt[wave][nn * 32 +  8 + h2 * 16];

        // --- out(16x64) += P(16x32) * V(32x64), four 16-col B tiles from LDS
        #pragma unroll
        for (int j = 0; j < 4; ++j) {
            const v16bf bv = *(const v16bf*)(Vs + (j * 16 + nn) * KT + h2 * 16);
            acc[j] = wmma_bf16(pa.v, bv, acc[j]);
        }

        __syncthreads();                        // protect buffers before reuse
    }

    // store tile to bf16 attention buffer [n*L + q][h*64 + d]
    const int n  = nh >> 3;
    const int hh = nh & 7;
    #pragma unroll
    for (int j = 0; j < 4; ++j) {
        #pragma unroll
        for (int r = 0; r < 8; ++r) {
            const int m = r + 8 * h2;
            Ab[((size_t)(n * LQ + q0 + m)) * FEA + hh * HD + j * 16 + nn] =
                (__bf16)acc[j][r];
        }
    }
}

// ---------------------------------------------------------------------------
// Kernel 4: final projection [8192 x 512] @ Wo[512 x 64] + bo, WMMA over 16
// K-steps. One 16x64 tile per wave.
// ---------------------------------------------------------------------------
__global__ __launch_bounds__(256)
void sa_out_kernel(const __bf16* __restrict__ Ab,
                   const __bf16* __restrict__ Wot,
                   const float* __restrict__ bo,
                   float* __restrict__ out) {
    const int wave = threadIdx.x >> 5;
    const int lane = threadIdx.x & 31;
    const int nn   = lane & 15;
    const int h2   = lane >> 4;
    const int row0 = (blockIdx.x * 8 + wave) * 16;

    v8f z = {};
    v8f acc[4] = {z, z, z, z};

    const __bf16* arow = Ab + (size_t)(row0 + nn) * FEA;
    for (int st = 0; st < FEA / 32; ++st) {
        const int kb = st * 32;
        V16U a;
        a.h[0] = *(const v8bf*)(arow + kb +      h2 * 16);
        a.h[1] = *(const v8bf*)(arow + kb +  8 + h2 * 16);
        #pragma unroll
        for (int j = 0; j < 4; ++j) {
            const v16bf b = *(const v16bf*)(Wot + (size_t)(j * 16 + nn) * FEA
                                            + kb + h2 * 16);
            acc[j] = wmma_bf16(a.v, b, acc[j]);
        }
    }

    #pragma unroll
    for (int j = 0; j < 4; ++j) {
        const int col = j * 16 + nn;
        const float bv = bo[col];
        #pragma unroll
        for (int r = 0; r < 8; ++r) {
            const int m = r + 8 * h2;
            out[(size_t)(row0 + m) * HD + col] = acc[j][r] + bv;
        }
    }
}

// ---------------------------------------------------------------------------
extern "C" void kernel_launch(void* const* d_in, const int* in_sizes, int n_in,
                              void* d_out, int out_size, void* d_ws, size_t ws_size,
                              hipStream_t stream) {
    const float* query = (const float*)d_in[0];
    const float* Wv    = (const float*)d_in[1];
    const float* Wk    = (const float*)d_in[2];
    const float* Wq    = (const float*)d_in[3];
    const float* Wo    = (const float*)d_in[4];
    const float* bo    = (const float*)d_in[5];
    float* out = (float*)d_out;

    char* ws = (char*)d_ws;
    const size_t MB8 = 8ull << 20;          // 4*8*2048*64 bf16 = 8 MB exactly
    __bf16* Qp  = (__bf16*)(ws + 0 * MB8);
    __bf16* Kp  = (__bf16*)(ws + 1 * MB8);
    __bf16* Vt  = (__bf16*)(ws + 2 * MB8);
    __bf16* Ab  = (__bf16*)(ws + 3 * MB8);  // 4*2048*512 bf16 = 8 MB
    __bf16* Wot = (__bf16*)(ws + 4 * MB8);  // 64 KB

    sa_proj_kernel<<<NB * LQ * NH, 64, 0, stream>>>(query, Wv, Wk, Wq, Qp, Kp, Vt);
    sa_wot_kernel<<<(FEA * HD) / 256, 256, 0, stream>>>(Wo, Wot);
    sa_attn_kernel<<<(NB * NH) * (LQ / 128), 256, 0, stream>>>(Qp, Kp, Vt, Ab);
    sa_out_kernel<<<(NB * LQ) / 128, 256, 0, stream>>>(Ab, Wot, bo, out);
}
